// Embedding_33449205301618
// MI455X (gfx1250) — compile-verified, matrix operand order checked by reading, not run on hardware
//
#include <hip/hip_runtime.h>
#include <hip/hip_bf16.h>

typedef __attribute__((ext_vector_type(16))) __bf16 v16bf;
typedef __attribute__((ext_vector_type(8)))  float  v8f;

#define DEV   192
#define PED   64
#define EMB   256
#define OUTW  512
#define XS_STRIDE 266   // 133 dwords: gcd(133 mod 64 = 5, 64) = 1 -> conflict-free
#define YS_STRIDE 522   // 261 dwords: gcd(261 mod 64 = 5, 64) = 1 -> conflict-free

__device__ __forceinline__ float lrelu(float x) { return x >= 0.f ? x : 0.01f * x; }

union BfPair { __bf16 b[2]; unsigned u; };
union AFrag  { v16bf v; unsigned u[8]; };

// ---------------------------------------------------------------------------
// Pack W [K,N] (f32 row-major) into bf16 WMMA B-fragment order:
// frag = kstep*(N/16)+ntile; element (lane,e): k = kstep*32 + ((e>>3)<<4) +
// ((lane>>4)<<3) + (e&7), n = ntile*16 + (lane&15)   (ISA 16-bit B layout)
// ---------------------------------------------------------------------------
__global__ void pack_w(const float* __restrict__ W, __bf16* __restrict__ Wp,
                       int K, int N) {
  int t = blockIdx.x * blockDim.x + threadIdx.x;
  if (t >= K * N) return;
  int e      = t & 15;
  int lane   = (t >> 4) & 31;
  int frag   = t >> 9;
  int ntiles = N >> 4;
  int kstep  = frag / ntiles;
  int ntile  = frag - kstep * ntiles;
  int k = (kstep << 5) + ((e >> 3) << 4) + ((lane >> 4) << 3) + (e & 7);
  int n = (ntile << 4) + (lane & 15);
  Wp[t] = (__bf16)W[(size_t)k * N + n];
}

// ---------------------------------------------------------------------------
// batch_index: searchsorted(state_index[1:], row, side='right')
// ---------------------------------------------------------------------------
__global__ void seg_index(const int* __restrict__ sidx, int nseg, int M,
                          int* __restrict__ bidx, int* __restrict__ out_bi) {
  int r = blockIdx.x * blockDim.x + threadIdx.x;
  if (r >= M) return;
  int lo = 0, hi = nseg;
  while (lo < hi) {
    int mid = (lo + hi) >> 1;
    if (sidx[mid + 1] <= r) lo = mid + 1; else hi = mid;
  }
  bidx[r] = lo;
  out_bi[r] = lo;
}

// ---------------------------------------------------------------------------
// GEMM1: h = lrelu([states|PE] @ W1 + b1)  -> out[:, 0:256]
// block = 8 waves = 32 rows x 256 cols; wave = 16 rows x 64 cols
// ---------------------------------------------------------------------------
__global__ __launch_bounds__(256)
void gemm1(const float* __restrict__ states, const int* __restrict__ sidx,
           const int* __restrict__ bidx, const __bf16* __restrict__ W1p,
           const float* __restrict__ b1, float* __restrict__ out) {
  __shared__ __bf16 xs[32 * XS_STRIDE];
  const int tid = threadIdx.x;
  const int mbase = blockIdx.x * 32;

  // ---- stage x-tile (states f32->bf16, PE computed in-flight) ----
  {
    int row  = tid >> 3;           // 0..31
    int cg   = tid & 7;            // 8 col-groups of 32
    int grow = mbase + row;
    __bf16* xr = &xs[row * XS_STRIDE];
    if (cg < 6) {                  // cols 0..191: states
      const float4* src = (const float4*)(states + (size_t)grow * DEV + cg * 32);
      #pragma unroll
      for (int j = 0; j < 8; ++j) {
        float4 v = src[j];
        BfPair p0; p0.b[0] = (__bf16)v.x; p0.b[1] = (__bf16)v.y;
        BfPair p1; p1.b[0] = (__bf16)v.z; p1.b[1] = (__bf16)v.w;
        int c = cg * 32 + j * 4;
        *(unsigned*)&xr[c]     = p0.u;
        *(unsigned*)&xr[c + 2] = p1.u;
      }
    } else {                       // cols 192..255: sinusoidal PE
      int b = bidx[grow];
      float pos = (float)(grow - sidx[b] + 1);
      int pbase = (cg - 6) * 32;
      #pragma unroll
      for (int j = 0; j < 16; ++j) {
        int p = pbase + 2 * j;     // even PE column
        int i = p >> 1;
        float invf = __expf(-0.28782313662425575f * (float)i); // 10000^(-i/32)
        float ang = pos * invf;
        BfPair pr; pr.b[0] = (__bf16)__sinf(ang); pr.b[1] = (__bf16)__cosf(ang);
        *(unsigned*)&xr[DEV + p] = pr.u;
      }
    }
  }
  __syncthreads();

  // ---- WMMA: 8 k-steps of 16x16x32 bf16, 4 N-subtiles per wave ----
  const int lane  = tid & 31;
  const int wave  = tid >> 5;
  const int g     = wave & 3;             // col group (64 cols)
  const int rbase = (wave >> 2) << 4;     // row half (16 rows)
  const int koff  = (lane >> 4) << 3;     // ISA A-layout K split
  const __bf16* axr = &xs[(rbase + (lane & 15)) * XS_STRIDE];

  v8f acc[4] = {};
  #pragma unroll
  for (int ks = 0; ks < 8; ++ks) {
    AFrag a;
    #pragma unroll
    for (int j = 0; j < 8; ++j) {
      int e = 2 * j;
      int k = (ks << 5) + ((e >> 3) << 4) + koff + (e & 7);
      a.u[j] = *(const unsigned*)&axr[k];       // ds_load_b32
    }
    const __bf16* wp = W1p + ((size_t)(ks * 16 + g * 4) * 32 + lane) * 16;
    #pragma unroll
    for (int t = 0; t < 4; ++t) {
      v16bf bf = *(const v16bf*)(wp + (size_t)t * 512);
      acc[t] = __builtin_amdgcn_wmma_f32_16x16x32_bf16(
          false, a.v, false, bf, (short)0, acc[t], false, false);
    }
  }

  // ---- epilogue: bias + LeakyReLU, h -> out[:, 0:256] ----
  const int mrow = mbase + rbase + ((lane >> 4) << 3);
  #pragma unroll
  for (int t = 0; t < 4; ++t) {
    int n = g * 64 + t * 16 + (lane & 15);
    float bias = b1[n];
    #pragma unroll
    for (int r = 0; r < 8; ++r)
      out[(size_t)(mrow + r) * OUTW + n] = lrelu(acc[t][r] + bias);
  }
}

// ---------------------------------------------------------------------------
// Segment mean+max over contiguous row ranges (no atomics; h is L2-hot)
// ---------------------------------------------------------------------------
__global__ __launch_bounds__(256)
void seg_reduce(const float* __restrict__ out, const int* __restrict__ sidx,
                float* __restrict__ red) {
  int b = blockIdx.x;
  int c = threadIdx.x;
  int s0 = sidx[b], s1 = sidx[b + 1];
  float sum = 0.f, mx = -3.402823466e38f;
  for (int r = s0; r < s1; ++r) {
    float v = out[(size_t)r * OUTW + c];
    sum += v;
    mx = fmaxf(mx, v);
  }
  int cnt = s1 - s0;
  float mean = sum / fmaxf((float)cnt, 1.f);
  if (cnt <= 0) mx = 0.f;
  red[(size_t)b * 512 + c]       = mean;
  red[(size_t)b * 512 + 256 + c] = mx;
}

// ---------------------------------------------------------------------------
// GEMM2: agg = lrelu([mean|max] @ W2 + b2)   (1024 x 512 @ 512 x 256)
// ---------------------------------------------------------------------------
__global__ __launch_bounds__(256)
void gemm2(const float* __restrict__ red, const __bf16* __restrict__ W2p,
           const float* __restrict__ b2, float* __restrict__ agg) {
  __shared__ __bf16 ys[32 * YS_STRIDE];
  const int tid = threadIdx.x;
  const int mbase = blockIdx.x * 32;
  {
    int row = tid >> 3;
    int cg  = tid & 7;                 // 8 groups of 64 cols
    const float4* src = (const float4*)(red + (size_t)(mbase + row) * 512 + cg * 64);
    __bf16* yr = &ys[row * YS_STRIDE + cg * 64];
    #pragma unroll
    for (int j = 0; j < 16; ++j) {
      float4 v = src[j];
      BfPair p0; p0.b[0] = (__bf16)v.x; p0.b[1] = (__bf16)v.y;
      BfPair p1; p1.b[0] = (__bf16)v.z; p1.b[1] = (__bf16)v.w;
      *(unsigned*)&yr[j * 4]     = p0.u;
      *(unsigned*)&yr[j * 4 + 2] = p1.u;
    }
  }
  __syncthreads();

  const int lane  = tid & 31;
  const int wave  = tid >> 5;
  const int g     = wave & 3;
  const int rbase = (wave >> 2) << 4;
  const int koff  = (lane >> 4) << 3;
  const __bf16* ayr = &ys[(rbase + (lane & 15)) * YS_STRIDE];

  v8f acc[4] = {};
  #pragma unroll
  for (int ks = 0; ks < 16; ++ks) {
    AFrag a;
    #pragma unroll
    for (int j = 0; j < 8; ++j) {
      int e = 2 * j;
      int k = (ks << 5) + ((e >> 3) << 4) + koff + (e & 7);
      a.u[j] = *(const unsigned*)&ayr[k];
    }
    const __bf16* wp = W2p + ((size_t)(ks * 16 + g * 4) * 32 + lane) * 16;
    #pragma unroll
    for (int t = 0; t < 4; ++t) {
      v16bf bf = *(const v16bf*)(wp + (size_t)t * 512);
      acc[t] = __builtin_amdgcn_wmma_f32_16x16x32_bf16(
          false, a.v, false, bf, (short)0, acc[t], false, false);
    }
  }
  const int mrow = mbase + rbase + ((lane >> 4) << 3);
  #pragma unroll
  for (int t = 0; t < 4; ++t) {
    int n = g * 64 + t * 16 + (lane & 15);
    float bias = b2[n];
    #pragma unroll
    for (int r = 0; r < 8; ++r)
      agg[(size_t)(mrow + r) * EMB + n] = lrelu(acc[t][r] + bias);
  }
}

// ---------------------------------------------------------------------------
// Scatter agg[batch_index] -> out[:, 256:512] (float4)
// ---------------------------------------------------------------------------
__global__ __launch_bounds__(256)
void scatter_agg(const float* __restrict__ agg, const int* __restrict__ bidx,
                 float* __restrict__ out, int M) {
  int t = blockIdx.x * blockDim.x + threadIdx.x;   // one float4 each
  int r  = t >> 6;
  int c4 = (t & 63) << 2;
  if (r >= M) return;
  int b = bidx[r];
  float4 v = *(const float4*)(agg + (size_t)b * EMB + c4);
  *(float4*)(out + (size_t)r * OUTW + EMB + c4) = v;
}

// ---------------------------------------------------------------------------
extern "C" void kernel_launch(void* const* d_in, const int* in_sizes, int n_in,
                              void* d_out, int out_size, void* d_ws, size_t ws_size,
                              hipStream_t stream) {
  const float* states = (const float*)d_in[0];
  const int*   sidx   = (const int*)d_in[1];
  const float* W1     = (const float*)d_in[2];
  const float* b1     = (const float*)d_in[3];
  const float* W2     = (const float*)d_in[4];
  const float* b2     = (const float*)d_in[5];
  const int M    = in_sizes[0] / DEV;       // 262144
  const int nseg = in_sizes[1] - 1;         // 1024

  float* out = (float*)d_out;
  int* out_bi = (int*)(out + (size_t)M * OUTW);   // batch_index tail of d_out

  char* ws = (char*)d_ws;
  int*    bidx = (int*)ws;                                       // M*4 B
  __bf16* W1p  = (__bf16*)(ws + (size_t)M * 4);                  // 128 KB
  __bf16* W2p  = (__bf16*)(ws + (size_t)M * 4 + 256 * 256 * 2);  // 256 KB
  float*  red  = (float*)(ws + (size_t)M * 4 + 256 * 256 * 2 + 512 * 256 * 2);
  float*  agg  = red + (size_t)nseg * 512;

  pack_w<<<(256 * 256 + 255) / 256, 256, 0, stream>>>(W1, W1p, 256, 256);
  pack_w<<<(512 * 256 + 255) / 256, 256, 0, stream>>>(W2, W2p, 512, 256);
  seg_index<<<(M + 255) / 256, 256, 0, stream>>>(sidx, nseg, M, bidx, out_bi);
  gemm1<<<M / 32, 256, 0, stream>>>(states, sidx, bidx, W1p, b1, out);
  seg_reduce<<<nseg, 256, 0, stream>>>(out, sidx, red);
  gemm2<<<nseg / 32, 256, 0, stream>>>(red, W2p, b2, agg);
  scatter_agg<<<(M * 64 + 255) / 256, 256, 0, stream>>>(agg, bidx, out, M);
}
